// TransformerCrossAttnLayerWithReturnAttn_77936476553401
// MI455X (gfx1250) — compile-verified, matrix operand order checked by reading, not run on hardware
//
#include <hip/hip_runtime.h>
#include <hip/hip_bf16.h>

// ---------------------------------------------------------------------------
// Types for CDNA5 WMMA (wave32): 16x16x32 bf16 -> f32 accumulate
// ---------------------------------------------------------------------------
typedef __attribute__((ext_vector_type(16))) __bf16 v16bf;
typedef __attribute__((ext_vector_type(8)))  float  v8f;
typedef __attribute__((ext_vector_type(4)))  unsigned int v4u;
typedef __attribute__((ext_vector_type(8)))  int    v8i_t;
typedef __attribute__((ext_vector_type(4)))  int    v4i_t;

__device__ __forceinline__ v8f wmma_bf16(v16bf a, v16bf b, v8f c) {
  // (neg_a, A, neg_b, B, c_mod, C, reuse_a, reuse_b)
  return __builtin_amdgcn_wmma_f32_16x16x32_bf16(false, a, false, b, (short)0, c,
                                                 false, false);
}

#define E_DIM 1024
#define H_DIM 16
#define DH    64
#define FF    4096
#define BQ    2048      // B*Lq = 4*512
#define BK    8192      // B*Lk = 4*2048

// ---------------------------------------------------------------------------
// fp32 -> bf16 cast (grid-stride)
// ---------------------------------------------------------------------------
__global__ void xattn_cast_bf16(const float* __restrict__ s,
                                __bf16* __restrict__ d, int n) {
  int i = blockIdx.x * blockDim.x + threadIdx.x;
  int stride = gridDim.x * blockDim.x;
  for (; i < n; i += stride) d[i] = (__bf16)s[i];
}

// ---------------------------------------------------------------------------
// Generic bf16 WMMA GEMM: out = A[MxK] @ W[NxK]^T + bias, optional ReLU.
// Workgroup: 256 threads (8 waves); WG tile 128(M) x 128(N);
// wave tile 32(M) x 64(N) = 8 accumulators -> 8 WMMAs per 6 fragment loads.
// A, W fragments loaded directly from global (all operands L2-resident).
// ---------------------------------------------------------------------------
__global__ __launch_bounds__(256)
void xattn_gemm_bf16(const __bf16* __restrict__ A, const __bf16* __restrict__ W,
                     const float* __restrict__ bias,
                     float* __restrict__ outF, __bf16* __restrict__ outB,
                     int M, int N, int K, int relu) {
  const int lane = threadIdx.x & 31;
  const int wave = threadIdx.x >> 5;
  const int m0 = blockIdx.y * 128 + (wave & 3) * 32;
  const int n0 = blockIdx.x * 128 + (wave >> 2) * 64;

  const int ar = lane & 15;          // row/col within fragment
  const int kh = (lane >> 4) << 4;   // which 16-wide K half this lane holds

  const __bf16* a0p = A + (long)(m0 + ar) * K + kh;
  const __bf16* a1p = a0p + 16L * K;
  const __bf16* bp[4];
  #pragma unroll
  for (int j = 0; j < 4; ++j)
    bp[j] = W + (long)(n0 + j * 16 + ar) * K + kh;

  v8f acc[2][4] = {};
  for (int k0 = 0; k0 < K; k0 += 32) {
    __builtin_prefetch((const void*)(a0p + k0 + 256), 0, 1);
    __builtin_prefetch((const void*)(bp[0] + k0 + 256), 0, 1);
    v16bf a0 = *(const v16bf*)(a0p + k0);
    v16bf a1 = *(const v16bf*)(a1p + k0);
    #pragma unroll
    for (int j = 0; j < 4; ++j) {
      v16bf b = *(const v16bf*)(bp[j] + k0);
      acc[0][j] = wmma_bf16(a0, b, acc[0][j]);
      acc[1][j] = wmma_bf16(a1, b, acc[1][j]);
    }
  }

  // C layout: VGPR r of lane l holds (M = r + 8*(l>>4), N = l&15) of the tile.
  const int rbase = (lane >> 4) * 8;
  #pragma unroll
  for (int i = 0; i < 2; ++i) {
    #pragma unroll
    for (int j = 0; j < 4; ++j) {
      int col = n0 + j * 16 + ar;
      float bv = bias ? bias[col] : 0.f;
      #pragma unroll
      for (int r = 0; r < 8; ++r) {
        int row = m0 + i * 16 + rbase + r;
        float v = acc[i][j][r] + bv;
        if (relu) v = v > 0.f ? v : 0.f;
        long idx = (long)row * N + col;
        if (outF) outF[idx] = v;
        if (outB) outB[idx] = (__bf16)v;
      }
    }
  }
}

// ---------------------------------------------------------------------------
// v[B*Lk, E] -> vt[B,H,Dh,Lk] so attn@V B-fragments are contiguous in K.
// ---------------------------------------------------------------------------
__global__ void xattn_transpose_v(const __bf16* __restrict__ v,
                                  __bf16* __restrict__ vt) {
  const long total = 4L * H_DIM * DH * 2048;
  long i = (long)blockIdx.x * blockDim.x + threadIdx.x;
  long stride = (long)gridDim.x * blockDim.x;
  for (; i < total; i += stride) {
    int k = (int)(i & 2047);
    long t = i >> 11;
    int d = (int)(t & 63); t >>= 6;
    int h = (int)(t & 15);
    int b = (int)(t >> 4);
    vt[i] = v[((long)(b * 2048 + k)) * E_DIM + h * DH + d];
  }
}

// ---------------------------------------------------------------------------
// Fused attention per (b, h, 64-row q block):
//   TDM: stage 64x64 Q tile into LDS via tensor_load_to_lds (TENSORcnt)
//   scores = (Q @ K^T)/8 (WMMA)  -> 64x2048 bf16 strip in LDS (256 KB)
//   softmax (wave shuffles)      -> fp32 attn to d_out + bf16 probs in LDS
//   ctx = P @ V (WMMA, A from LDS)
// LDS: scores 64*2048*2 = 262144 B, qtile 64*64*2 = 8192 B  (270336 total)
// ---------------------------------------------------------------------------
__global__ __launch_bounds__(256)
void xattn_attention(const __bf16* __restrict__ Q,   // [B*512, 1024]
                     const __bf16* __restrict__ Kp,  // [B*2048, 1024]
                     const __bf16* __restrict__ VT,  // [B*H*64, 2048]
                     float* __restrict__ attn,       // [B*H*512, 2048]
                     __bf16* __restrict__ ctx) {     // [B*512, 1024]
  extern __shared__ __bf16 sm[];                     // scores [64][2048]
  __bf16* qtile = sm + 64 * 2048;                    // [64][64]
  const int lane = threadIdx.x & 31;
  const int wave = threadIdx.x >> 5;
  const int qb = blockIdx.x & 7;
  const int h  = (blockIdx.x >> 3) & 15;
  const int b  = blockIdx.x >> 7;
  const int q0 = qb * 64;

  const int ar = lane & 15;
  const int kh = (lane >> 4) << 4;

  const __bf16* qbase = Q  + ((long)(b * 512 + q0)) * E_DIM + h * DH;
  const __bf16* kbase = Kp + ((long)b * 2048) * E_DIM + h * DH;

  // ---- phase 0: DMA the 64x64 Q tile (row stride E_DIM) into LDS -------
#if __has_builtin(__builtin_amdgcn_tensor_load_to_lds) && \
    __has_builtin(__builtin_amdgcn_s_wait_tensorcnt)
  if (wave == 0) {
    unsigned lds_off = (unsigned)(uintptr_t)(void*)qtile;  // low 32 = LDS byte offset
    unsigned long long ga = (unsigned long long)(uintptr_t)qbase;
    // D# group 0: count=1 | lds_addr | global_addr[56:0] | type=2
    v4u g0;
    g0.x = 1u;
    g0.y = lds_off;
    g0.z = (unsigned)(ga & 0xffffffffu);
    g0.w = (unsigned)((ga >> 32) & 0x01ffffffu) | (2u << 30);
    // D# group 1: data_size=2B; tensor 64x64 (dim0 stride 1024); tile 64x64
    v8i_t g1;
    g1[0] = (1 << 16);          // workgroup_mask=0, data_size=1 (2 bytes)
    g1[1] = (64 << 16);         // tensor_dim0[15:0] at bits[63:48]
    g1[2] = (64 << 16);         // tensor_dim0 hi=0, tensor_dim1[15:0]
    g1[3] = (64 << 16);         // tensor_dim1 hi=0, tile_dim0=64
    g1[4] = 64;                 // tile_dim1=64, tile_dim2=0
    g1[5] = E_DIM;              // tensor_dim0_stride = 1024 elements
    g1[6] = 0;                  // stride hi, tensor_dim1_stride lo
    g1[7] = 0;
    v4i_t gz4 = {0, 0, 0, 0};
    v8i_t gz8 = {0, 0, 0, 0, 0, 0, 0, 0};
    __builtin_amdgcn_tensor_load_to_lds(g0, g1, gz4, gz4, gz8, 0);
    __builtin_amdgcn_s_wait_tensorcnt(0);
  }
#else
  // fallback: cooperative load of the Q tile
  for (int i = threadIdx.x; i < 64 * 64; i += 256)
    qtile[i] = qbase[(long)(i >> 6) * E_DIM + (i & 63)];
#endif
  __syncthreads();

  // ---- phase 1: scores (A from LDS Q tile, B rows from global K) -------
  {
    const int qsub  = wave & 3;        // 16-row q subtile
    const int khalf = wave >> 2;       // which half of the 128 k tiles
    const __bf16* ap = qtile + (qsub * 16 + ar) * DH + kh;
    v16bf a_lo = *(const v16bf*)ap;
    v16bf a_hi = *(const v16bf*)(ap + 32);
    const int rb = (lane >> 4) * 8;
    for (int kt = khalf * 64; kt < khalf * 64 + 64; ++kt) {
      const __bf16* bp = kbase + (long)(kt * 16 + ar) * E_DIM + kh;
      v8f c = {};
      c = wmma_bf16(a_lo, *(const v16bf*)bp,        c);
      c = wmma_bf16(a_hi, *(const v16bf*)(bp + 32), c);
      int col = kt * 16 + ar;
      #pragma unroll
      for (int r = 0; r < 8; ++r)
        sm[(qsub * 16 + rb + r) * 2048 + col] = (__bf16)(c[r] * 0.125f);
    }
  }
  __syncthreads();

  // ---- phase 2: softmax (one wave per row, 8 rows per wave) ------------
  {
    float* attnBase = attn + ((long)((b * H_DIM + h) * 512 + q0)) * 2048;
    for (int row = wave; row < 64; row += 8) {
      __bf16* srow = sm + row * 2048;
      float mx = -3.0e38f;
      for (int c = lane; c < 2048; c += 32)
        mx = fmaxf(mx, (float)srow[c]);
      for (int off = 16; off; off >>= 1)
        mx = fmaxf(mx, __shfl_xor(mx, off, 32));
      float s = 0.f;
      for (int c = lane; c < 2048; c += 32)
        s += __expf((float)srow[c] - mx);
      for (int off = 16; off; off >>= 1)
        s += __shfl_xor(s, off, 32);
      float inv = 1.f / s;
      for (int c = lane; c < 2048; c += 32) {
        float p = __expf((float)srow[c] - mx) * inv;
        attnBase[(long)row * 2048 + c] = p;   // fp32 attn output
        srow[c] = (__bf16)p;                  // bf16 probs for P@V
      }
    }
  }
  __syncthreads();

  // ---- phase 3: ctx = P @ V (A from LDS, B from VT) --------------------
  {
    const __bf16* vbase = VT + ((long)((b * H_DIM + h) * DH)) * 2048;
    const int qs  = wave >> 1;
    const int dt0 = (wave & 1) * 2;
    v8f acc0 = {}, acc1 = {};
    const __bf16* arow = sm + (qs * 16 + ar) * 2048 + kh;
    const __bf16* b0p = vbase + (long)(dt0 * 16 + ar) * 2048 + kh;
    const __bf16* b1p = b0p + 16L * 2048;
    for (int k0 = 0; k0 < 2048; k0 += 32) {
      v16bf a  = *(const v16bf*)(arow + k0);
      v16bf b0 = *(const v16bf*)(b0p + k0);
      v16bf b1 = *(const v16bf*)(b1p + k0);
      acc0 = wmma_bf16(a, b0, acc0);
      acc1 = wmma_bf16(a, b1, acc1);
    }
    const int rb = (lane >> 4) * 8;
    __bf16* cbase = ctx + ((long)(b * 512 + q0)) * E_DIM + h * DH;
    #pragma unroll
    for (int j = 0; j < 2; ++j) {
      v8f& c = j ? acc1 : acc0;
      int d = (dt0 + j) * 16 + ar;
      #pragma unroll
      for (int r = 0; r < 8; ++r)
        cbase[(long)(qs * 16 + rb + r) * E_DIM + d] = (__bf16)c[r];
    }
  }
}

// ---------------------------------------------------------------------------
// y = LayerNorm(x1 + x2) * g + be ; one block per row of 1024.
// ---------------------------------------------------------------------------
__global__ __launch_bounds__(256)
void xattn_add_ln(const float* __restrict__ x1, const float* __restrict__ x2,
                  const float* __restrict__ g, const float* __restrict__ be,
                  float* __restrict__ outF, __bf16* __restrict__ outB) {
  const int row = blockIdx.x;
  const float* a  = x1 + (long)row * E_DIM;
  const float* bb = x2 + (long)row * E_DIM;
  const int lane = threadIdx.x & 31;
  const int wave = threadIdx.x >> 5;
  __shared__ float red[8];

  float v[4];
  float s = 0.f;
  #pragma unroll
  for (int i = 0; i < 4; ++i) {
    int col = threadIdx.x + i * 256;
    v[i] = a[col] + bb[col];
    s += v[i];
  }
  for (int off = 16; off; off >>= 1) s += __shfl_xor(s, off, 32);
  if (lane == 0) red[wave] = s;
  __syncthreads();
  float tot = 0.f;
  #pragma unroll
  for (int i = 0; i < 8; ++i) tot += red[i];
  const float mu = tot * (1.f / E_DIM);
  __syncthreads();

  float ss = 0.f;
  #pragma unroll
  for (int i = 0; i < 4; ++i) { float d = v[i] - mu; ss += d * d; }
  for (int off = 16; off; off >>= 1) ss += __shfl_xor(ss, off, 32);
  if (lane == 0) red[wave] = ss;
  __syncthreads();
  float tot2 = 0.f;
  #pragma unroll
  for (int i = 0; i < 8; ++i) tot2 += red[i];
  const float rstd = rsqrtf(tot2 * (1.f / E_DIM) + 1e-5f);

  #pragma unroll
  for (int i = 0; i < 4; ++i) {
    int col = threadIdx.x + i * 256;
    float y = (v[i] - mu) * rstd * g[col] + be[col];
    long idx = (long)row * E_DIM + col;
    if (outF) outF[idx] = y;
    if (outB) outB[idx] = (__bf16)y;
  }
}

// ---------------------------------------------------------------------------
// Host launcher
// ---------------------------------------------------------------------------
extern "C" void kernel_launch(void* const* d_in, const int* in_sizes, int n_in,
                              void* d_out, int out_size, void* d_ws, size_t ws_size,
                              hipStream_t stream) {
  (void)in_sizes; (void)n_in; (void)out_size; (void)ws_size;

  const float* query = (const float*)d_in[0];
  const float* keyv  = (const float*)d_in[1];
  const float* wq = (const float*)d_in[2];
  const float* wk = (const float*)d_in[3];
  const float* wv = (const float*)d_in[4];
  const float* bq = (const float*)d_in[5];
  const float* bk = (const float*)d_in[6];
  const float* bv = (const float*)d_in[7];
  const float* wo = (const float*)d_in[8];
  const float* bo = (const float*)d_in[9];
  const float* g1  = (const float*)d_in[10];
  const float* be1 = (const float*)d_in[11];
  const float* w1  = (const float*)d_in[12];
  const float* b1  = (const float*)d_in[13];
  const float* w2  = (const float*)d_in[14];
  const float* b2  = (const float*)d_in[15];
  const float* g2  = (const float*)d_in[16];
  const float* be2 = (const float*)d_in[17];

  float* q3_out   = (float*)d_out;                       // [2048,1024]
  float* attn_out = (float*)d_out + (long)BQ * E_DIM;    // [4,16,512,2048]

  // scratch layout
  char* base = (char*)d_ws;
  size_t off = 0;
  auto alloc = [&](size_t bytes) -> char* {
    char* p = base + off;
    off += (bytes + 255) & ~(size_t)255;
    return p;
  };
  __bf16* qx_bf = (__bf16*)alloc((size_t)BQ * E_DIM * 2);
  __bf16* kv_bf = (__bf16*)alloc((size_t)BK * E_DIM * 2);
  __bf16* wq_bf = (__bf16*)alloc((size_t)E_DIM * E_DIM * 2);
  __bf16* wk_bf = (__bf16*)alloc((size_t)E_DIM * E_DIM * 2);
  __bf16* wv_bf = (__bf16*)alloc((size_t)E_DIM * E_DIM * 2);
  __bf16* wo_bf = (__bf16*)alloc((size_t)E_DIM * E_DIM * 2);
  __bf16* w1_bf = (__bf16*)alloc((size_t)FF * E_DIM * 2);
  __bf16* w2_bf = (__bf16*)alloc((size_t)E_DIM * FF * 2);
  __bf16* q_bf  = (__bf16*)alloc((size_t)BQ * E_DIM * 2);
  __bf16* k_bf  = (__bf16*)alloc((size_t)BK * E_DIM * 2);
  __bf16* v_bf  = (__bf16*)alloc((size_t)BK * E_DIM * 2);
  __bf16* vt_bf = (__bf16*)alloc((size_t)BK * E_DIM * 2);
  __bf16* ctx_bf= (__bf16*)alloc((size_t)BQ * E_DIM * 2);
  float*  ao_f  = (float*) alloc((size_t)BQ * E_DIM * 4);
  float*  q2_f  = (float*) alloc((size_t)BQ * E_DIM * 4);
  __bf16* q2_bf = (__bf16*)alloc((size_t)BQ * E_DIM * 2);
  __bf16* h_bf  = (__bf16*)alloc((size_t)BQ * FF * 2);
  float*  ffn_f = (float*) alloc((size_t)BQ * E_DIM * 4);

  // 1) casts to bf16
  auto cast = [&](const float* s, __bf16* d, int n) {
    xattn_cast_bf16<<<dim3(4096), dim3(256), 0, stream>>>(s, d, n);
  };
  cast(query, qx_bf, BQ * E_DIM);
  cast(keyv,  kv_bf, BK * E_DIM);
  cast(wq, wq_bf, E_DIM * E_DIM);
  cast(wk, wk_bf, E_DIM * E_DIM);
  cast(wv, wv_bf, E_DIM * E_DIM);
  cast(wo, wo_bf, E_DIM * E_DIM);
  cast(w1, w1_bf, FF * E_DIM);
  cast(w2, w2_bf, E_DIM * FF);

  // 2) QKV projections (WMMA GEMM); WG tile 128x128
  xattn_gemm_bf16<<<dim3(E_DIM / 128, BQ / 128), dim3(256), 0, stream>>>(
      qx_bf, wq_bf, bq, nullptr, q_bf, BQ, E_DIM, E_DIM, 0);
  xattn_gemm_bf16<<<dim3(E_DIM / 128, BK / 128), dim3(256), 0, stream>>>(
      kv_bf, wk_bf, bk, nullptr, k_bf, BK, E_DIM, E_DIM, 0);
  xattn_gemm_bf16<<<dim3(E_DIM / 128, BK / 128), dim3(256), 0, stream>>>(
      kv_bf, wv_bf, bv, nullptr, v_bf, BK, E_DIM, E_DIM, 0);

  // 3) transpose V to [B,H,Dh,Lk]
  xattn_transpose_v<<<dim3(8192), dim3(256), 0, stream>>>(v_bf, vt_bf);

  // 4) fused attention (TDM Q tile + scores + softmax + P@V); 264 KB LDS
  static const size_t attn_lds = (64 * 2048 + 64 * 64) * sizeof(__bf16);
  (void)hipFuncSetAttribute((const void*)xattn_attention,
                            hipFuncAttributeMaxDynamicSharedMemorySize,
                            (int)attn_lds);
  xattn_attention<<<dim3(512), dim3(256), attn_lds, stream>>>(
      q_bf, k_bf, vt_bf, attn_out, ctx_bf);

  // 5) output projection -> fp32
  xattn_gemm_bf16<<<dim3(E_DIM / 128, BQ / 128), dim3(256), 0, stream>>>(
      ctx_bf, wo_bf, bo, ao_f, nullptr, BQ, E_DIM, E_DIM, 0);

  // 6) residual + LN1
  xattn_add_ln<<<dim3(BQ), dim3(256), 0, stream>>>(query, ao_f, g1, be1, q2_f, q2_bf);

  // 7) FFN
  xattn_gemm_bf16<<<dim3(FF / 128, BQ / 128), dim3(256), 0, stream>>>(
      q2_bf, w1_bf, b1, nullptr, h_bf, BQ, FF, E_DIM, 1);
  xattn_gemm_bf16<<<dim3(E_DIM / 128, BQ / 128), dim3(256), 0, stream>>>(
      h_bf, w2_bf, b2, ffn_f, nullptr, BQ, E_DIM, FF, 0);

  // 8) residual + LN2 -> q3
  xattn_add_ln<<<dim3(BQ), dim3(256), 0, stream>>>(q2_f, ffn_f, g2, be2, q3_out, nullptr);
}
